// ChildSumTreeLSTMCell_24730421691110
// MI455X (gfx1250) — matrix operand order repaired and checked by reading, not compiled, not run on hardware
//
#include <hip/hip_runtime.h>
#include <stdint.h>

#define DD 128          // embedding dim
#define HH 128          // hidden dim
#define KTOT 256        // D + H (GEMM K)
#define APAD 8          // LDS row pad (uint16 units) -> 16B; row stride 132 dwords -> conflict-free b128 reads
#define MROWS 32        // node rows per workgroup (2 M-tiles, B-fragment reused 2x)

typedef __attribute__((ext_vector_type(16))) __bf16 v16bf;
typedef __attribute__((ext_vector_type(8)))  float  v8f;

union ABfrag { uint32_t u[8]; uint4 q[2]; v16bf v; };

__device__ __forceinline__ uint16_t f2bf(float f) {
  union { float f; uint32_t u; } x; x.f = f;
  uint32_t u = x.u;
  u += 0x7FFFu + ((u >> 16) & 1u);      // round-to-nearest-even
  return (uint16_t)(u >> 16);
}

// ---------------- zero workspace (h_sum, c_sum) ----------------
__global__ void __launch_bounds__(256)
k_zero(float4* __restrict__ p, long n4) {
  long i = (long)blockIdx.x * blockDim.x + threadIdx.x;
  if (i < n4) p[i] = make_float4(0.f, 0.f, 0.f, 0.f);
}

// ---------------- pack weights into WMMA B-fragment order ----------------
// frag = ct*8 + kc (ct: col-tile of 16 in 0..31, kc: K-chunk of 32 in 0..7)
// dword index in frag = lane*8 + v ; n=(lane&15), kbase=(lane<16)?0:16,
// dword v holds bf16 pair K = kc*32 + kbase + 2v (lo), +1 (hi)
__global__ void __launch_bounds__(256)
k_prep(const float* __restrict__ Wf, const float* __restrict__ Uf,
       const float* __restrict__ Wiou, const float* __restrict__ Uiou,
       uint32_t* __restrict__ Bfrag) {
  int t = blockIdx.x * blockDim.x + threadIdx.x;   // 0..65535
  if (t >= 32 * 8 * 256) return;
  int frag = t >> 8;
  int idx  = t & 255;
  int lane = idx >> 3, v = idx & 7;
  int ct = frag >> 3, kc = frag & 7;
  int n = ct * 16 + (lane & 15);                   // global out-column 0..511
  int kbase = (lane < 16) ? 0 : 16;
  uint32_t pk = 0;
  #pragma unroll
  for (int h = 0; h < 2; ++h) {
    int k = kc * 32 + kbase + 2 * v + h;           // 0..255
    float val;
    if (n < HH) {                                  // forget-gate columns
      val = (k < DD) ? Wf[k * HH + n] : Uf[(k - DD) * HH + n];
    } else {                                       // iou columns
      int cc = n - HH;
      val = (k < DD) ? Wiou[k * (3 * HH) + cc] : Uiou[(k - DD) * (3 * HH) + cc];
    }
    pk |= ((uint32_t)f2bf(val)) << (16 * h);
  }
  Bfrag[t] = pk;
}

// ---------------- edge scatter: h_sum/c_sum segment sums ----------------
// one wave per edge; lane handles 4 consecutive columns
__global__ void __launch_bounds__(256)
k_scatter(const float* __restrict__ h, const float* __restrict__ c,
          const int* __restrict__ child, const int* __restrict__ parent,
          float* __restrict__ hsum, float* __restrict__ csum, int E) {
  long tid = (long)blockIdx.x * blockDim.x + threadIdx.x;
  if (tid >= (long)E * 32) return;
  int e  = (int)(tid >> 5);
  int c0 = (int)(tid & 31) * 4;
  int ch = child[e], pa = parent[e];
  float4 hv = *(const float4*)(h + (size_t)ch * HH + c0);
  float4 cv = *(const float4*)(c + (size_t)ch * HH + c0);
  float* hp = hsum + (size_t)pa * HH + c0;
  float* cp = csum + (size_t)pa * HH + c0;
  unsafeAtomicAdd(hp + 0, hv.x); unsafeAtomicAdd(hp + 1, hv.y);
  unsafeAtomicAdd(hp + 2, hv.z); unsafeAtomicAdd(hp + 3, hv.w);
  unsafeAtomicAdd(cp + 0, cv.x); unsafeAtomicAdd(cp + 1, cv.y);
  unsafeAtomicAdd(cp + 2, cv.z); unsafeAtomicAdd(cp + 3, cv.w);
}

// ---------------- fused GEMM (bf16 WMMA) + LSTM gating ----------------
// 32 node rows per workgroup; wave w owns h-columns [16w,16w+16) and computes
// all 4 gate tiles for 2 M-tiles, so each B fragment feeds 2 WMMAs.
__global__ void __launch_bounds__(256)
k_gemm(const float* __restrict__ x, const float* __restrict__ hsum,
       const float* __restrict__ csum, const float* __restrict__ b_f,
       const float* __restrict__ b_iou, const uint32_t* __restrict__ Bfrag,
       float* __restrict__ out, int N) {
  __shared__ __align__(16) uint16_t Abuf[MROWS][KTOT + APAD];
  const int tid  = threadIdx.x;
  const int row0 = blockIdx.x * MROWS;

  // stage A = [x | h_sum] (32 x 256) as bf16 into LDS, coalesced float4 loads
  #pragma unroll
  for (int i = 0; i < 8; ++i) {
    int f4 = tid + i * 256;                // 0..2047 float4 slots
    int r  = f4 >> 6;                      // row 0..31 (64 float4 per row)
    int cq = (f4 & 63) << 2;               // col 0..252 step 4
    int node = row0 + r; if (node >= N) node = N - 1;
    const float* src = (cq < DD) ? (x + (size_t)node * DD + cq)
                                 : (hsum + (size_t)node * HH + (cq - DD));
    float4 v = *(const float4*)src;
    uint16_t* dst = &Abuf[r][cq];
    dst[0] = f2bf(v.x); dst[1] = f2bf(v.y); dst[2] = f2bf(v.z); dst[3] = f2bf(v.w);
  }
  __syncthreads();

  const int wave = tid >> 5;               // 0..7  -> h-columns [16w, 16w+16)
  const int lane = tid & 31;
  const int mrow = lane & 15;              // A row within M-tile
  const int akb  = (lane < 16) ? 0 : 16;   // A K-base within chunk (contiguous K, matches Bfrag)

  v8f acc[2][4];                           // [mtile][gate]  gate: 0:f 1:i 2:o 3:u
  #pragma unroll
  for (int mt = 0; mt < 2; ++mt)
    #pragma unroll
    for (int t = 0; t < 4; ++t) acc[mt][t] = (v8f){0,0,0,0,0,0,0,0};

  #pragma unroll
  for (int kc = 0; kc < 8; ++kc) {
    ABfrag a0, a1;
    const uint4* ap0 = (const uint4*)&Abuf[mrow][kc * 32 + akb];       // 16B aligned
    const uint4* ap1 = (const uint4*)&Abuf[mrow + 16][kc * 32 + akb];
    a0.q[0] = ap0[0]; a0.q[1] = ap0[1];     // 16 contiguous halfwords
    a1.q[0] = ap1[0]; a1.q[1] = ap1[1];
    #pragma unroll
    for (int t = 0; t < 4; ++t) {
      int ct = t * 8 + wave;                // gate t, columns 16w..16w+15
      ABfrag b;
      const uint4* bp = (const uint4*)(Bfrag + (((size_t)(ct * 8 + kc)) << 8) + (lane << 3));
      b.q[0] = bp[0]; b.q[1] = bp[1];
      acc[0][t] = __builtin_amdgcn_wmma_f32_16x16x32_bf16(
                      false, a0.v, false, b.v, (short)0, acc[0][t], false, false);
      acc[1][t] = __builtin_amdgcn_wmma_f32_16x16x32_bf16(
                      false, a1.v, false, b.v, (short)0, acc[1][t], false, false);
    }
  }

  // epilogue: bias + LSTM gating; C/D layout: VGPR r, lanes0-15 M=r, lanes16-31 M=r+8, N=lane&15
  const int hcol = wave * 16 + (lane & 15);
  const float bf_ = b_f[hcol];
  const float bi  = b_iou[hcol];
  const float bo  = b_iou[HH + hcol];
  const float bu  = b_iou[2 * HH + hcol];

  #pragma unroll
  for (int mt = 0; mt < 2; ++mt) {
    #pragma unroll
    for (int r = 0; r < 8; ++r) {
      int m    = mt * 16 + r + ((lane >> 4) << 3);
      int node = row0 + m;
      bool ok  = (node < N);
      int nd   = ok ? node : (N - 1);
      float cs = csum[(size_t)nd * HH + hcol];
      float pf = acc[mt][0][r] + bf_;
      float pi = acc[mt][1][r] + bi;
      float po = acc[mt][2][r] + bo;
      float pu = acc[mt][3][r] + bu;
      float fg = 1.f / (1.f + __expf(-pf));
      float ig = 1.f / (1.f + __expf(-pi));
      float og = 1.f / (1.f + __expf(-po));
      float cn = ig * tanhf(pu) + fg * cs;
      float hn = og * tanhf(cn);
      if (ok) {
        out[(size_t)node * (2 * HH) + hcol]      = hn;
        out[(size_t)node * (2 * HH) + HH + hcol] = cn;
      }
    }
  }
}

extern "C" void kernel_launch(void* const* d_in, const int* in_sizes, int n_in,
                              void* d_out, int out_size, void* d_ws, size_t ws_size,
                              hipStream_t stream) {
  const float* x      = (const float*)d_in[0];
  const float* h      = (const float*)d_in[1];
  const float* c      = (const float*)d_in[2];
  const int*   child  = (const int*)d_in[3];
  const int*   parent = (const int*)d_in[4];
  const float* W_f    = (const float*)d_in[5];
  const float* U_f    = (const float*)d_in[6];
  const float* b_f    = (const float*)d_in[7];
  const float* W_iou  = (const float*)d_in[8];
  const float* U_iou  = (const float*)d_in[9];
  const float* b_iou  = (const float*)d_in[10];
  float* out = (float*)d_out;

  const int N = in_sizes[0] / DD;
  const int E = in_sizes[3];

  float*    hsum  = (float*)d_ws;
  float*    csum  = hsum + (size_t)N * HH;
  uint32_t* Bfrag = (uint32_t*)(csum + (size_t)N * HH);

  long n4 = ((long)2 * N * HH) / 4;
  k_zero<<<(int)((n4 + 255) / 256), 256, 0, stream>>>((float4*)d_ws, n4);
  k_prep<<<256, 256, 0, stream>>>(W_f, U_f, W_iou, U_iou, Bfrag);
  long st = (long)E * 32;
  k_scatter<<<(int)((st + 255) / 256), 256, 0, stream>>>(h, c, child, parent, hsum, csum, E);
  k_gemm<<<(N + MROWS - 1) / MROWS, 256, 0, stream>>>(x, hsum, csum, b_f, b_iou, Bfrag, out, N);
}
